// GLCFormer_47708496724359
// MI455X (gfx1250) — matrix operand order repaired and checked by reading, not compiled, not run on hardware
//
#include <hip/hip_runtime.h>

typedef _Float16 f16;
typedef __attribute__((ext_vector_type(16))) _Float16 v16h;
typedef __attribute__((ext_vector_type(8)))  _Float16 v8h;
typedef __attribute__((ext_vector_type(8)))  float    v8f;

#define DIM    128
#define WT     64
#define NW     1344
#define NBATCH 4
#define QBLK   21
#define RSCALE 0.08838834764831845f   // 128^-0.5

// ---- f16 workspace layout (offsets in halves) ----
#define H_MLOCT  0          // 128x128  transposed (s*Wql@Wkl^T)^T
#define H_MGLOBT 16384      // 128x128  transposed (s*Wq@Wk^T)^T
#define H_WVLT   32768      // Wvl^T
#define H_WVT    49152      // Wv^T
#define H_WGLOT  65536      // gate_w[:128]^T
#define H_WGHIT  81920      // gate_w[128:]^T
#define H_WC0    98304      // conv0 as [512][9*128] f16
#define H_WC1    688128     // conv1 as [512][9*128] f16
#define H_CENTH  1277952    // centers f16 [4][1344][128]
#define H_TOTAL  1966080
#define F_BASE_BYTES (H_TOTAL*2)
// ---- f32 workspace (offsets in floats, after F_BASE_BYTES) ----
#define F_CENTF  0          // centers f32
#define F_YCENT  688128     // y_centers f32
#define F_GA     1376256    // y_centers @ gate_w[:128]

#define OUT1_OFF 2097152
#define OUT2_OFF 10485760

// ============================================================
// CDNA5 async global->LDS DMA (ASYNCcnt) + wait. Each active lane
// copies 16B from its global address to its LDS address.
// ============================================================
__device__ __forceinline__ void async_copy_b128(void* lds, const void* gptr) {
    unsigned int       l  = (unsigned int)(unsigned long long)(size_t)lds; // LDS addr = low 32b of flat
    unsigned long long ga = (unsigned long long)(size_t)gptr;
    asm volatile("global_load_async_to_lds_b128 %0, %1, off"
                 :: "v"(l), "v"(ga) : "memory");
}
__device__ __forceinline__ void wait_async0() {
    asm volatile("s_wait_asynccnt 0" ::: "memory");
}

// ============================================================
// WMMA fragment loaders (CDNA5 wave32 16x16x32 f16 layouts)
// A (16x32): lane L holds row M=L%16; halves j: K = k0 + j + 8*((j>>3)+(L>>4))
//   -> two contiguous 16B runs at k0+8h and k0+16+8h.
// B (32x16): lane L holds col N=L%16; halves j: K = k0 + j + 8*((j>>3)+(L>>4))
//   -> two contiguous 16B runs at k0+8h and k0+16+8h; source array must be N-major ("transposed").
// ============================================================
__device__ __forceinline__ v16h load_a(const f16* base, int rs, int m0, int k0, int lane) {
    const f16* p = base + (size_t)(m0 + (lane & 15)) * rs + k0 + ((lane >> 4) << 3);
    v8h lo = *(const v8h*)p;
    v8h hi = *(const v8h*)(p + 16);
    v16h r;
#pragma unroll
    for (int j = 0; j < 8; ++j) { r[j] = lo[j]; r[j + 8] = hi[j]; }
    return r;
}

__device__ __forceinline__ v16h load_bt(const f16* baseT, int rs, int n0, int k0, int lane) {
    const f16* p = baseT + (size_t)(n0 + (lane & 15)) * rs + k0 + ((lane >> 4) << 3);
    v8h lo = *(const v8h*)p;
    v8h hi = *(const v8h*)(p + 16);
    v16h r;
#pragma unroll
    for (int j = 0; j < 8; ++j) { r[j] = lo[j]; r[j + 8] = hi[j]; }
    return r;
}

__device__ __forceinline__ v8f wmma_f16(v16h a, v16h b, v8f c) {
    return __builtin_amdgcn_wmma_f32_16x16x32_f16(false, a, false, b, (short)0, c, false, false);
}

// ============================================================
// Kernel 1a: attention/gate weight prep (transpose + fold Q/K)
// grid 128 x block 128
// ============================================================
__global__ void prep_attn_weights(const float* Wq, const float* Wk,
                                  const float* Wql, const float* Wkl,
                                  const float* Wv, const float* Wvl,
                                  const float* gate_w, f16* wsH) {
    int n = blockIdx.x;   // output row of transposed matrices
    int k = threadIdx.x;
    float sg = 0.f, sl = 0.f;
    for (int j = 0; j < DIM; ++j) {
        sg += Wq[k * DIM + j] * Wk[n * DIM + j];
        sl += Wql[k * DIM + j] * Wkl[n * DIM + j];
    }
    wsH[H_MGLOBT + n * DIM + k] = (f16)(sg * RSCALE);
    wsH[H_MLOCT  + n * DIM + k] = (f16)(sl * RSCALE);
    wsH[H_WVT    + n * DIM + k] = (f16)Wv[k * DIM + n];
    wsH[H_WVLT   + n * DIM + k] = (f16)Wvl[k * DIM + n];
    wsH[H_WGLOT  + n * DIM + k] = (f16)gate_w[k * DIM + n];
    wsH[H_WGHIT  + n * DIM + k] = (f16)gate_w[(DIM + k) * DIM + n];
}

// ============================================================
// Kernel 1b: conv weight repack  [O][I][3][3] -> f16 [O][(dy*3+dx)*128+i]
// ============================================================
__global__ void prep_conv_weights(const float* conv0, const float* conv1, f16* wsH) {
    int idx = blockIdx.x * blockDim.x + threadIdx.x;
    if (idx >= 512 * 1152) return;
    int o = idx / 1152, r = idx % 1152;
    int pos = r / DIM, i = r % DIM;
    wsH[H_WC0 + idx] = (f16)conv0[(o * DIM + i) * 9 + pos];
    wsH[H_WC1 + idx] = (f16)conv1[(o * DIM + i) * 9 + pos];
}

// ============================================================
// Kernel 2: window means (centers), f32 + f16 copies
// grid B*NW x block 128 (thread = channel)
// ============================================================
__global__ void centers_kernel(const float* f0, const float* f1, const float* f2,
                               float* centf, f16* centh) {
    int blk = blockIdx.x;
    int b = blk / NW, w = blk % NW;
    int c = threadIdx.x;
    const float* src; int HW, p0;
    if (w < 64)       { src = f0; HW = 4096;  p0 = w * WT; }
    else if (w < 320) { src = f1; HW = 16384; p0 = (w - 64) * WT; }
    else              { src = f2; HW = 65536; p0 = (w - 320) * WT; }
    const float* p = src + (size_t)(b * DIM + c) * HW + p0;
    float s = 0.f;
#pragma unroll
    for (int t = 0; t < WT; ++t) s += p[t];
    s *= (1.0f / 64.0f);
    centf[(size_t)blk * DIM + c] = s;
    centh[(size_t)blk * DIM + c] = (f16)s;
}

// ============================================================
// Kernel 3: global path. grid = B*QBLK (one 64-query block each), 256 thr.
// Tq = C_blk @ Mglob; two-pass softmax over 1344 keys; U = P @ C;
// y = (U @ Wv) * C;  gA = y @ gate_lo.
// Key blocks are staged into LDS via async DMA (ASYNCcnt).
// ============================================================
__global__ __launch_bounds__(256)
void global_attn_kernel(const f16* wsH, const float* centf, float* ycent, float* gA) {
    __shared__ __align__(16) f16 sTq[64 * DIM];   // Tq, later U, later y (f16)
    __shared__ __align__(16) f16 sC[64 * DIM];    // key block (row-major), async-staged
    __shared__ __align__(16) f16 sCt[DIM * 72];   // key block, channel-major
    __shared__ __align__(16) f16 sP[64 * 72];     // normalized probabilities
    __shared__ float sPM[2][64], sPL[2][64], sM[64], sL[64];

    const int tid = threadIdx.x, lane = tid & 31, wv = tid >> 5;
    const int b = blockIdx.x / QBLK, qb = blockIdx.x % QBLK;
    const int q0 = qb * 64;
    const f16* centh = wsH + H_CENTH + (size_t)b * NW * DIM;

    // ---- Tq = centers[q0:q0+64] @ Mglob (K=128), 4 tiles/wave ----
    const int rU = wv >> 1, cU = (wv & 1) * 4;    // 64x128 tiling, 4 tiles/wave
    {
        const f16* mgT = wsH + H_MGLOBT;
        v8f acc[4] = {};
#pragma unroll
        for (int ks = 0; ks < 4; ++ks) {
            v16h a = load_a(centh, DIM, q0 + rU * 16, ks * 32, lane);
#pragma unroll
            for (int q = 0; q < 4; ++q) {
                v16h bf = load_bt(mgT, DIM, (cU + q) * 16, ks * 32, lane);
                acc[q] = wmma_f16(a, bf, acc[q]);
            }
        }
#pragma unroll
        for (int q = 0; q < 4; ++q)
#pragma unroll
            for (int r = 0; r < 8; ++r) {
                int m = rU * 16 + r + ((lane >> 4) << 3);
                int n = (cU + q) * 16 + (lane & 15);
                sTq[m * DIM + n] = (f16)acc[q][r];
            }
    }
    if (tid < 64) { sM[tid] = -1e30f; sL[tid] = 0.f; }
    __syncthreads();

    // S tiling: 2 tiles/wave (rS = wv&3, cols (wv>>2)*2 + {0,1})
    const int rS = wv & 3, cS = (wv >> 2) * 2;

    // ---- pass 1: running row max / sum over 21 key blocks ----
    for (int j = 0; j < QBLK; ++j) {
        // async DMA key block j (16KB f16) into sC; prefetch block j+1
        for (int t = tid; t < 1024; t += 256)
            async_copy_b128(sC + t * 8, centh + (size_t)j * 64 * DIM + t * 8);
        if (j + 1 < QBLK)
            __builtin_prefetch(centh + (size_t)(j + 1) * 64 * DIM + tid * 64, 0, 1);
        wait_async0();
        __syncthreads();

        v8f s0 = {}, s1 = {};
#pragma unroll
        for (int ks = 0; ks < 4; ++ks) {
            v16h a  = load_a(sTq, DIM, rS * 16, ks * 32, lane);
            v16h b0 = load_bt(sC, DIM, cS * 16, ks * 32, lane);
            v16h b1 = load_bt(sC, DIM, (cS + 1) * 16, ks * 32, lane);
            s0 = wmma_f16(a, b0, s0);
            s1 = wmma_f16(a, b1, s1);
        }
#pragma unroll
        for (int r = 0; r < 8; ++r) {
            float pm = fmaxf(s0[r], s1[r]);
#pragma unroll
            for (int msk = 1; msk < 16; msk <<= 1) pm = fmaxf(pm, __shfl_xor(pm, msk, 32));
            if ((lane & 15) == 0) sPM[wv >> 2][rS * 16 + r + ((lane >> 4) << 3)] = pm;
        }
        __syncthreads();
#pragma unroll
        for (int r = 0; r < 8; ++r) {
            int row = rS * 16 + r + ((lane >> 4) << 3);
            float mb = fmaxf(sPM[0][row], sPM[1][row]);
            float ps = __expf(s0[r] - mb) + __expf(s1[r] - mb);
#pragma unroll
            for (int msk = 1; msk < 16; msk <<= 1) ps += __shfl_xor(ps, msk, 32);
            if ((lane & 15) == 0) sPL[wv >> 2][row] = ps;
        }
        __syncthreads();
        if (tid < 64) {
            float mb = fmaxf(sPM[0][tid], sPM[1][tid]);
            float lb = sPL[0][tid] + sPL[1][tid];
            float M = sM[tid], Mn = fmaxf(M, mb);
            sL[tid] = sL[tid] * __expf(M - Mn) + lb * __expf(mb - Mn);
            sM[tid] = Mn;
        }
        __syncthreads();
    }

    // ---- pass 2: U = sum_j P_j @ C_j ----
    v8f U[4] = {};
    for (int j = 0; j < QBLK; ++j) {
        for (int t = tid; t < 1024; t += 256)
            async_copy_b128(sC + t * 8, centh + (size_t)j * 64 * DIM + t * 8);
        if (j + 1 < QBLK)
            __builtin_prefetch(centh + (size_t)(j + 1) * 64 * DIM + tid * 64, 0, 1);
        wait_async0();
        __syncthreads();
        // build channel-major copy (LDS -> LDS)
        for (int t = tid; t < 64 * DIM; t += 256) {
            int key = t >> 7, cch = t & 127;
            sCt[cch * 72 + key] = sC[key * DIM + cch];
        }
        __syncthreads();
        v8f s0 = {}, s1 = {};
#pragma unroll
        for (int ks = 0; ks < 4; ++ks) {
            v16h a  = load_a(sTq, DIM, rS * 16, ks * 32, lane);
            v16h b0 = load_bt(sC, DIM, cS * 16, ks * 32, lane);
            v16h b1 = load_bt(sC, DIM, (cS + 1) * 16, ks * 32, lane);
            s0 = wmma_f16(a, b0, s0);
            s1 = wmma_f16(a, b1, s1);
        }
#pragma unroll
        for (int r = 0; r < 8; ++r) {
            int row = rS * 16 + r + ((lane >> 4) << 3);
            float m = sM[row], li = 1.0f / sL[row];
            sP[row * 72 + cS * 16 + (lane & 15)]       = (f16)(__expf(s0[r] - m) * li);
            sP[row * 72 + (cS + 1) * 16 + (lane & 15)] = (f16)(__expf(s1[r] - m) * li);
        }
        __syncthreads();
#pragma unroll
        for (int ks = 0; ks < 2; ++ks) {
            v16h a = load_a(sP, 72, rU * 16, ks * 32, lane);
#pragma unroll
            for (int q = 0; q < 4; ++q) {
                v16h bf = load_bt(sCt, 72, (cU + q) * 16, ks * 32, lane);
                U[q] = wmma_f16(a, bf, U[q]);
            }
        }
        __syncthreads();
    }

    // ---- store U f16 -> sTq ----
#pragma unroll
    for (int q = 0; q < 4; ++q)
#pragma unroll
        for (int r = 0; r < 8; ++r) {
            int m = rU * 16 + r + ((lane >> 4) << 3);
            int n = (cU + q) * 16 + (lane & 15);
            sTq[m * DIM + n] = (f16)U[q][r];
        }
    __syncthreads();

    // ---- y = (U @ Wv) * centers ----
    const float* cfB = centf + (size_t)b * NW * DIM;
    float* ycB = ycent + (size_t)b * NW * DIM;
    float* gaB = gA + (size_t)b * NW * DIM;
    const f16* wvT = wsH + H_WVT;
    v8f Y[4] = {};
#pragma unroll
    for (int ks = 0; ks < 4; ++ks) {
        v16h a = load_a(sTq, DIM, rU * 16, ks * 32, lane);
#pragma unroll
        for (int q = 0; q < 4; ++q) {
            v16h bf = load_bt(wvT, DIM, (cU + q) * 16, ks * 32, lane);
            Y[q] = wmma_f16(a, bf, Y[q]);
        }
    }
#pragma unroll
    for (int q = 0; q < 4; ++q)
#pragma unroll
        for (int r = 0; r < 8; ++r) {
            int m = rU * 16 + r + ((lane >> 4) << 3);
            int n = (cU + q) * 16 + (lane & 15);
            float yv = Y[q][r] * cfB[(size_t)(q0 + m) * DIM + n];
            Y[q][r] = yv;
            ycB[(size_t)(q0 + m) * DIM + n] = yv;
        }
    __syncthreads();
#pragma unroll
    for (int q = 0; q < 4; ++q)
#pragma unroll
        for (int r = 0; r < 8; ++r) {
            int m = rU * 16 + r + ((lane >> 4) << 3);
            int n = (cU + q) * 16 + (lane & 15);
            sTq[m * DIM + n] = (f16)Y[q][r];
        }
    __syncthreads();

    // ---- gA = y @ gate_w[:128] ----
    const f16* wgLoT = wsH + H_WGLOT;
    v8f G[4] = {};
#pragma unroll
    for (int ks = 0; ks < 4; ++ks) {
        v16h a = load_a(sTq, DIM, rU * 16, ks * 32, lane);
#pragma unroll
        for (int q = 0; q < 4; ++q) {
            v16h bf = load_bt(wgLoT, DIM, (cU + q) * 16, ks * 32, lane);
            G[q] = wmma_f16(a, bf, G[q]);
        }
    }
#pragma unroll
    for (int q = 0; q < 4; ++q)
#pragma unroll
        for (int r = 0; r < 8; ++r) {
            int m = rU * 16 + r + ((lane >> 4) << 3);
            int n = (cU + q) * 16 + (lane & 15);
            gaB[(size_t)(q0 + m) * DIM + n] = G[q][r];
        }
}

// ============================================================
// Kernel 4: fused local attention + gated fusion per window.
// grid = B*NW, block 256.  T=X@Mloc; S=T@X^T; softmax; P2=S@X;
// Yl=(P2@Wvl)*X; alpha=sigmoid(gA + Yl@Wg_hi); y -> maps in d_out.
// ============================================================
__global__ __launch_bounds__(256)
void local_fused_kernel(const float* f0, const float* f1, const float* f2,
                        const f16* wsH, const float* ycent, const float* gA,
                        float* out) {
    __shared__ __align__(16) f16 sX[64 * DIM];    // window tokens (row-major)
    __shared__ __align__(16) f16 sXt[DIM * 72];   // channel-major copy
    __shared__ __align__(16) f16 sT[64 * DIM];    // T, then P2, then Yl (f16)
    __shared__ __align__(16) f16 sS[64 * 72];     // softmaxed scores
    __shared__ float sPM[2][64], sPL[2][64];

    int blk = blockIdx.x;
    int b = blk / NW, w = blk % NW;
    const float* src; int HW, p0; size_t obase;
    if (w < 64)       { src = f0; HW = 4096;  p0 = w * WT;         obase = 0; }
    else if (w < 320) { src = f1; HW = 16384; p0 = (w - 64) * WT;  obase = OUT1_OFF; }
    else              { src = f2; HW = 65536; p0 = (w - 320) * WT; obase = OUT2_OFF; }

    const int tid = threadIdx.x, lane = tid & 31, wv = tid >> 5;

    // warm caches for the three weight panels this block will stream
    __builtin_prefetch(wsH + H_MLOCT + tid * 64, 0, 2);
    __builtin_prefetch(wsH + H_WVLT  + tid * 64, 0, 2);
    __builtin_prefetch(wsH + H_WGHIT + tid * 64, 0, 2);

    for (int t = tid; t < WT * DIM; t += 256) {
        int tk = t & 63, c = t >> 6;
        f16 h = (f16)src[(size_t)(b * DIM + c) * HW + p0 + tk];
        sX[tk * DIM + c] = h;
        sXt[c * 72 + tk] = h;
    }
    __syncthreads();

    const int rU = wv >> 1, cU = (wv & 1) * 4;    // 64x128 tiling, 4 tiles/wave
    // ---- T = X @ Mloc ----
    {
        const f16* mT = wsH + H_MLOCT;
        v8f acc[4] = {};
#pragma unroll
        for (int ks = 0; ks < 4; ++ks) {
            v16h a = load_a(sX, DIM, rU * 16, ks * 32, lane);
#pragma unroll
            for (int q = 0; q < 4; ++q) {
                v16h bf = load_bt(mT, DIM, (cU + q) * 16, ks * 32, lane);
                acc[q] = wmma_f16(a, bf, acc[q]);
            }
        }
#pragma unroll
        for (int q = 0; q < 4; ++q)
#pragma unroll
            for (int r = 0; r < 8; ++r) {
                int m = rU * 16 + r + ((lane >> 4) << 3);
                int n = (cU + q) * 16 + (lane & 15);
                sT[m * DIM + n] = (f16)acc[q][r];
            }
    }
    __syncthreads();

    // ---- S = T @ X^T (64x64, K=128), 2 tiles/wave ----
    const int rS = wv & 3, cS = (wv >> 2) * 2;
    v8f s0 = {}, s1 = {};
#pragma unroll
    for (int ks = 0; ks < 4; ++ks) {
        v16h a  = load_a(sT, DIM, rS * 16, ks * 32, lane);
        v16h b0 = load_bt(sX, DIM, cS * 16, ks * 32, lane);
        v16h b1 = load_bt(sX, DIM, (cS + 1) * 16, ks * 32, lane);
        s0 = wmma_f16(a, b0, s0);
        s1 = wmma_f16(a, b1, s1);
    }
    // ---- exact softmax over 64 cols ----
#pragma unroll
    for (int r = 0; r < 8; ++r) {
        float pm = fmaxf(s0[r], s1[r]);
#pragma unroll
        for (int msk = 1; msk < 16; msk <<= 1) pm = fmaxf(pm, __shfl_xor(pm, msk, 32));
        if ((lane & 15) == 0) sPM[wv >> 2][rS * 16 + r + ((lane >> 4) << 3)] = pm;
    }
    __syncthreads();
#pragma unroll
    for (int r = 0; r < 8; ++r) {
        int row = rS * 16 + r + ((lane >> 4) << 3);
        float mb = fmaxf(sPM[0][row], sPM[1][row]);
        s0[r] = __expf(s0[r] - mb);
        s1[r] = __expf(s1[r] - mb);
        float ps = s0[r] + s1[r];
#pragma unroll
        for (int msk = 1; msk < 16; msk <<= 1) ps += __shfl_xor(ps, msk, 32);
        if ((lane & 15) == 0) sPL[wv >> 2][row] = ps;
    }
    __syncthreads();
#pragma unroll
    for (int r = 0; r < 8; ++r) {
        int row = rS * 16 + r + ((lane >> 4) << 3);
        float li = 1.0f / (sPL[0][row] + sPL[1][row]);
        sS[row * 72 + cS * 16 + (lane & 15)]       = (f16)(s0[r] * li);
        sS[row * 72 + (cS + 1) * 16 + (lane & 15)] = (f16)(s1[r] * li);
    }
    __syncthreads();

    // ---- P2 = S @ X (K=64) -> sT ----
    {
        v8f acc[4] = {};
#pragma unroll
        for (int ks = 0; ks < 2; ++ks) {
            v16h a = load_a(sS, 72, rU * 16, ks * 32, lane);
#pragma unroll
            for (int q = 0; q < 4; ++q) {
                v16h bf = load_bt(sXt, 72, (cU + q) * 16, ks * 32, lane);
                acc[q] = wmma_f16(a, bf, acc[q]);
            }
        }
#pragma unroll
        for (int q = 0; q < 4; ++q)
#pragma unroll
            for (int r = 0; r < 8; ++r) {
                int m = rU * 16 + r + ((lane >> 4) << 3);
                int n = (cU + q) * 16 + (lane & 15);
                sT[m * DIM + n] = (f16)acc[q][r];
            }
    }
    __syncthreads();

    // ---- Yl = (P2 @ Wvl) * X ----
    v8f YL[4] = {};
    {
        const f16* wvlT = wsH + H_WVLT;
#pragma unroll
        for (int ks = 0; ks < 4; ++ks) {
            v16h a = load_a(sT, DIM, rU * 16, ks * 32, lane);
#pragma unroll
            for (int q = 0; q < 4; ++q) {
                v16h bf = load_bt(wvlT, DIM, (cU + q) * 16, ks * 32, lane);
                YL[q] = wmma_f16(a, bf, YL[q]);
            }
        }
#pragma unroll
        for (int q = 0; q < 4; ++q)
#pragma unroll
            for (int r = 0; r < 8; ++r) {
                int m = rU * 16 + r + ((lane >> 4) << 3);
                int n = (cU + q) * 16 + (lane & 15);
                YL[q][r] *= (float)sX[m * DIM + n];
            }
    }
    __syncthreads();
#pragma unroll
    for (int q = 0; q < 4; ++q)
#pragma unroll
        for (int r = 0; r < 8; ++r) {
            int m = rU * 16 + r + ((lane >> 4) << 3);
            int n = (cU + q) * 16 + (lane & 15);
            sT[m * DIM + n] = (f16)YL[q][r];
        }
    __syncthreads();

    // ---- alphaB = Yl @ gate_w[128:] ----
    v8f AB[4] = {};
    {
        const f16* wgHiT = wsH + H_WGHIT;
#pragma unroll
        for (int ks = 0; ks < 4; ++ks) {
            v16h a = load_a(sT, DIM, rU * 16, ks * 32, lane);
#pragma unroll
            for (int q = 0; q < 4; ++q) {
                v16h bf = load_bt(wgHiT, DIM, (cU + q) * 16, ks * 32, lane);
                AB[q] = wmma_f16(a, bf, AB[q]);
            }
        }
    }

    // ---- fuse + write map ----
    const float* ycw = ycent + (size_t)(b * NW + w) * DIM;
    const float* gaw = gA    + (size_t)(b * NW + w) * DIM;
#pragma unroll
    for (int q = 0; q < 4; ++q) {
        int n = (cU + q) * 16 + (lane & 15);
        float yg = ycw[n];
        float ga = gaw[n];
#pragma unroll
        for (int r = 0; r < 8; ++r) {
            int m = rU * 16 + r + ((lane >> 4) << 3);
            float al = 1.0f / (1.0f + __expf(-(ga + AB[q][r])));
            float yv = al * yg + (1.0f - al) * YL[q][r];
            out[obase + (size_t)(b * DIM + n) * HW + p0 + m] = yv;
        }
    }
}

// ============================================================
// Kernel 5: implicit-GEMM conv3x3 + pixel-shuffle(2) + add.
// C-tile: 64 out-channels x 64 pixels of one input row; K = 9*128.
// grid = B * Hin * colBlocks * 8, block 256.
// ============================================================
__global__ __launch_bounds__(256)
void conv_ps_add_kernel(const float* x, const f16* wT, float* out,
                        int Hin, int colBlocks) {
    __shared__ __align__(16) f16 sB[64 * 40];     // im2col chunk [px][32 k]
    int blk = blockIdx.x;
    int ob = blk & 7; blk >>= 3;
    int cb = blk % colBlocks; blk /= colBlocks;
    int y = blk % Hin; int b = blk / Hin;
    const int W = Hin, px0 = cb * 64;
    const int tid = threadIdx.x, lane = tid & 31, wv = tid >> 5;
    const int rT = wv & 3, cT = (wv >> 2) * 2;

    // warm cache for this block's 64x1152 f16 weight panel
    __builtin_prefetch(wT + (size_t)(ob * 64) * 1152 + (size_t)tid * 576, 0, 2);

    v8f a0 = {}, a1 = {};
    for (int pos = 0; pos < 9; ++pos) {
        int dy = pos / 3 - 1, dx = pos % 3 - 1;
        int gy = y + dy;
        bool rowok = (gy >= 0 && gy < Hin);
        for (int i0 = 0; i0 < DIM; i0 += 32) {
            for (int t = tid; t < 32 * 64; t += 256) {
                int ii = t >> 6, px = t & 63;
                int gx = px0 + px + dx;
                float v = 0.f;
                if (rowok && gx >= 0 && gx < W)
                    v = x[(size_t)(b * DIM + i0 + ii) * Hin * W + (size_t)gy * W + gx];
                sB[px * 40 + ii] = (f16)v;
            }
            __syncthreads();
            int kg = pos * DIM + i0;
            v16h a  = load_a(wT, 1152, ob * 64 + rT * 16, kg, lane);
            v16h b0 = load_bt(sB, 40, cT * 16, 0, lane);
            v16h b1 = load_bt(sB, 40, (cT + 1) * 16, 0, lane);
            a0 = wmma_f16(a, b0, a0);
            a1 = wmma_f16(a, b1, a1);
            __syncthreads();
        }
    }
    // pixel shuffle epilogue: o = c4*4 + sy*2 + sx -> out[c4][2y+sy][2px+sx] +=
    int Hout = 2 * Hin, Wout = 2 * W;
#pragma unroll
    for (int q = 0; q < 2; ++q) {
        v8f& A = q ? a1 : a0;
        int px = px0 + (cT + q) * 16 + (lane & 15);
#pragma unroll
        for (int r = 0; r < 8; ++r) {
            int o = ob * 64 + rT * 16 + r + ((lane >> 4) << 3);
            int c4 = o >> 2, sy = (o >> 1) & 1, sx = o & 1;
            size_t idx = (size_t)(b * DIM + c4) * Hout * Wout
                       + (size_t)(2 * y + sy) * Wout + (2 * px + sx);
            out[idx] += A[r];
        }
    }
}

// ============================================================
extern "C" void kernel_launch(void* const* d_in, const int* in_sizes, int n_in,
                              void* d_out, int out_size, void* d_ws, size_t ws_size,
                              hipStream_t stream) {
    const float* f0    = (const float*)d_in[0];
    const float* f1    = (const float*)d_in[1];
    const float* f2    = (const float*)d_in[2];
    const float* Wq    = (const float*)d_in[3];
    const float* Wk    = (const float*)d_in[4];
    const float* Wv    = (const float*)d_in[5];
    const float* Wql   = (const float*)d_in[6];
    const float* Wkl   = (const float*)d_in[7];
    const float* Wvl   = (const float*)d_in[8];
    const float* gatew = (const float*)d_in[9];
    const float* conv0 = (const float*)d_in[10];
    const float* conv1 = (const float*)d_in[11];

    f16*   wsH   = (f16*)d_ws;
    float* wsF   = (float*)((char*)d_ws + F_BASE_BYTES);
    float* centf = wsF + F_CENTF;
    f16*   centh = wsH + H_CENTH;
    float* ycent = wsF + F_YCENT;
    float* gAbuf = wsF + F_GA;
    float* out   = (float*)d_out;

    prep_attn_weights<<<128, 128, 0, stream>>>(Wq, Wk, Wql, Wkl, Wv, Wvl, gatew, wsH);
    prep_conv_weights<<<(512 * 1152 + 255) / 256, 256, 0, stream>>>(conv0, conv1, wsH);
    centers_kernel<<<NBATCH * NW, 128, 0, stream>>>(f0, f1, f2, centf, centh);
    global_attn_kernel<<<NBATCH * QBLK, 256, 0, stream>>>(wsH, centf, ycent, gAbuf);
    local_fused_kernel<<<NBATCH * NW, 256, 0, stream>>>(f0, f1, f2, wsH, ycent, gAbuf, out);
    conv_ps_add_kernel<<<NBATCH * 64 * 1 * 8, 256, 0, stream>>>(out, wsH + H_WC0,
                                                                out + OUT1_OFF, 64, 1);
    conv_ps_add_kernel<<<NBATCH * 128 * 2 * 8, 256, 0, stream>>>(out + OUT1_OFF, wsH + H_WC1,
                                                                 out + OUT2_OFF, 128, 2);
}
